// DIN_64364379898509
// MI455X (gfx1250) — matrix operand order repaired and checked by reading, not compiled, not run on hardware
//
#include <hip/hip_runtime.h>

typedef __attribute__((ext_vector_type(16))) _Float16 v16h;
typedef __attribute__((ext_vector_type(8)))  _Float16 v8h;
typedef __attribute__((ext_vector_type(8)))  float    v8f;

#define B_   2048
#define T_   200
#define D_   64
#define H1_  80
#define H2_  40
#define K1M  128   // merged feature dim: [k, q*k]
#define K2P  96    // H1 padded to 3*32 for WMMA K
#define N2P  48    // H2 padded to 3*16 for WMMA N
#define NT_  13    // ceil(208/16) row tiles of T
#define NW_  4     // waves per block
#define CONST_MIN -4294967295.0f

// ---------------------------------------------------------------------------
// Prep (one tiny block, runs once):
//   w1m  : f16 col-major [80][128] merged W1 for per-tile GEMM:
//          rows 0..63  = W1b - W1c  (k coefficient)
//          rows 64..127= W1d        (q*k coefficient)
//   w2h  : f16 col-major zero-padded [48][96] W2
//   w1ac : f32 row-major [64][80] = W1a + W1c (q coefficient, folded per-block)
// ---------------------------------------------------------------------------
__global__ __launch_bounds__(256) void prep_kernel(const float* __restrict__ W1,
                                                   const float* __restrict__ W2,
                                                   _Float16* __restrict__ w1m,
                                                   _Float16* __restrict__ w2h,
                                                   float* __restrict__ w1ac) {
    const int tid = threadIdx.x;
    for (int i = tid; i < H1_ * K1M; i += 256) {
        const int n  = i >> 7;       // output column 0..79
        const int kd = i & 127;      // merged feature row 0..127
        float val;
        if (kd < 64) val = W1[(64 + kd) * H1_ + n] - W1[(128 + kd) * H1_ + n];
        else         val = W1[(192 + (kd - 64)) * H1_ + n];
        w1m[i] = (_Float16)val;
    }
    for (int i = tid; i < N2P * K2P; i += 256) {
        const int n  = i / K2P;      // padded output column 0..47
        const int kd = i - n * K2P;  // padded K 0..95
        const float val = (n < H2_ && kd < H1_) ? W2[kd * H2_ + n] : 0.f;
        w2h[i] = (_Float16)val;
    }
    // w1ac[d][n] = W1[d][n] + W1[128+d][n]  (identical row-major layout)
    for (int i = tid; i < D_ * H1_; i += 256)
        w1ac[i] = W1[i] + W1[128 * H1_ + i];
}

__device__ __forceinline__ void st4h(_Float16* p, float a, float b, float c, float d) {
    p[0] = (_Float16)a; p[1] = (_Float16)b; p[2] = (_Float16)c; p[3] = (_Float16)d;
}

__device__ __forceinline__ float sigmoidf_(float x) {
    return 1.f / (1.f + __expf(-x));
}

// ---------------------------------------------------------------------------
// Main kernel: one block (4 wave32) per batch element.
// ---------------------------------------------------------------------------
__global__ __launch_bounds__(128) void attn_mlp_kernel(
    const float* __restrict__ q,  const float* __restrict__ k,
    const float* __restrict__ v,  const int*   __restrict__ mask,
    const float* __restrict__ b1, const float* __restrict__ b2,
    const float* __restrict__ Wf, const float* __restrict__ bfp,
    const _Float16* __restrict__ w1m, const _Float16* __restrict__ w2h,
    const float* __restrict__ w1ac,
    float* __restrict__ out) {

    __shared__ __align__(16) _Float16 w1_sh[H1_ * K1M];     // col-major [80][128]
    __shared__ __align__(16) _Float16 w2_sh[N2P * K2P];     // col-major padded [48][96]
    __shared__ __align__(16) _Float16 a_sh [NW_][16 * K1M]; // per-wave A: 16 x 128
    __shared__ __align__(16) _Float16 h1_sh[NW_][16 * K2P]; // per-wave h1 f16 (padded K)
    __shared__ __align__(16) float    h2_sh[NW_][16 * N2P]; // per-wave h2 f32
    __shared__ __align__(16) float    q_sh[D_];
    __shared__ float pre_sh[H1_];                           // q@(W1a+W1c) + b1
    __shared__ float b2_sh[H2_], wf_sh[H2_];
    __shared__ float logits_sh[NT_ * 16];
    __shared__ float red_sh[128];
    __shared__ float part_sh[2][D_];

    const int tid  = threadIdx.x;
    const int w    = tid >> 5;          // wave id 0..3
    const int lane = tid & 31;          // wave32 lane
    const int b    = blockIdx.x;
    const int hi   = lane >> 4;         // lane half (0/1) for WMMA fragments
    const int lm   = lane & 15;         // row/col within 16

    // ---- stage weights / constants into LDS (coalesced b128 copies) ----
    {
        const uint4* s1 = (const uint4*)w1m; uint4* d1 = (uint4*)w1_sh;
        for (int i = tid; i < (H1_ * K1M) / 8; i += 128) d1[i] = s1[i];
        const uint4* s2 = (const uint4*)w2h; uint4* d2 = (uint4*)w2_sh;
        for (int i = tid; i < (N2P * K2P) / 8; i += 128) d2[i] = s2[i];
    }
    if (tid < D_) q_sh[tid] = q[(size_t)b * D_ + tid];
    for (int i = tid; i < H2_; i += 128) { b2_sh[i] = b2[i]; wf_sh[i] = Wf[i]; }
    const float bfv = bfp[0];
    __syncthreads();

    // ---- per-block fold: pre[n] = b1[n] + sum_d q[d] * w1ac[d][n] ----
    if (tid < H1_) {
        float s = b1[tid];
        #pragma unroll 8
        for (int d = 0; d < D_; ++d) s += q_sh[d] * w1ac[d * H1_ + tid];
        pre_sh[tid] = s;
    }
    // per-wave one-time init: zero h1 K-padding cols 80..95
    #pragma unroll
    for (int i = 0; i < 8; ++i) {
        const int idx = lane + i * 32;
        const int r = idx >> 4, c = H1_ + (idx & 15);
        h1_sh[w][r * K2P + c] = (_Float16)0.f;
    }
    __syncthreads();

    // ---- tile loop: each wave owns row-tiles tile = w, w+4, ... ----
    for (int tile = w; tile < NT_; tile += NW_) {
        const int t0 = tile * 16;

        // build merged features [k, q*k] (16 rows x 128) for this tile
        #pragma unroll
        for (int i = 0; i < 8; ++i) {
            const int idx = lane + i * 32;               // 0..255
            const int r = idx >> 4, c4 = (idx & 15) * 4; // d = c4..c4+3
            const int t = t0 + r;
            float4 kv;
            if (t < T_) {
                kv = *(const float4*)&k[((size_t)b * T_ + t) * D_ + c4];
                if (t + 16 * NW_ < T_)                   // prefetch next tile's rows
                    __builtin_prefetch(&k[((size_t)b * T_ + t + 16 * NW_) * D_ + c4], 0, 1);
            } else {
                kv = make_float4(0.f, 0.f, 0.f, 0.f);
            }
            const float4 qv = *(const float4*)&q_sh[c4];
            _Float16* row = &a_sh[w][r * K1M];
            st4h(row +      c4, kv.x, kv.y, kv.z, kv.w);                              // k
            st4h(row + 64 + c4, qv.x * kv.x, qv.y * kv.y, qv.z * kv.z, qv.w * kv.w);  // q*k
        }

        // ---- GEMM1: [16x128] x [128x80], acc pre-seeded with q-fold + b1 ----
        v8f acc[5];
        #pragma unroll
        for (int n = 0; n < 5; ++n) {
            const float p = pre_sh[n * 16 + lm];         // same for every row
            #pragma unroll
            for (int g = 0; g < 8; ++g) acc[n][g] = p;
        }
        #pragma unroll
        for (int kk = 0; kk < 4; ++kk) {
            const int K0 = kk * 32;
            union { v16h v; v8h h[2]; } af;              // A frag: row lm, contiguous 16 K
            const _Float16* ap = &a_sh[w][lm * K1M + K0 + hi * 16];
            af.h[0] = *(const v8h*)ap;
            af.h[1] = *(const v8h*)(ap + 8);
            #pragma unroll
            for (int n = 0; n < 5; ++n) {                // B frag: col lm, contiguous 16 K
                const v16h bv = *(const v16h*)&w1_sh[(n * 16 + lm) * K1M + K0 + hi * 16];
                acc[n] = __builtin_amdgcn_wmma_f32_16x16x32_f16(
                    false, af.v, false, bv, (short)0, acc[n], false, false);
            }
        }
        // sigmoid -> h1 (f16, padded-K layout); bias already folded into pre
        #pragma unroll
        for (int n = 0; n < 5; ++n) {
            const int col = n * 16 + lm;
            #pragma unroll
            for (int g = 0; g < 8; ++g) {
                const float s = sigmoidf_(acc[n][g]);
                h1_sh[w][(g + hi * 8) * K2P + col] = (_Float16)s;
            }
        }

        // ---- GEMM2: [16x96] x [96x48] (H1->H2 padded) ----
        v8f acc2[3];
        #pragma unroll
        for (int n = 0; n < 3; ++n) acc2[n] = (v8f){0.f,0.f,0.f,0.f,0.f,0.f,0.f,0.f};
        #pragma unroll
        for (int kk = 0; kk < 3; ++kk) {
            const int K0 = kk * 32;
            union { v16h v; v8h h[2]; } af;
            const _Float16* ap = &h1_sh[w][lm * K2P + K0 + hi * 16];
            af.h[0] = *(const v8h*)ap;
            af.h[1] = *(const v8h*)(ap + 8);
            #pragma unroll
            for (int n = 0; n < 3; ++n) {
                const v16h bv = *(const v16h*)&w2_sh[(n * 16 + lm) * K2P + K0 + hi * 16];
                acc2[n] = __builtin_amdgcn_wmma_f32_16x16x32_f16(
                    false, af.v, false, bv, (short)0, acc2[n], false, false);
            }
        }
        // bias + sigmoid -> h2 (f32)
        #pragma unroll
        for (int n = 0; n < 3; ++n) {
            const int col = n * 16 + lm;
            const float bias = (col < H2_) ? b2_sh[col] : 0.f;
            #pragma unroll
            for (int g = 0; g < 8; ++g)
                h2_sh[w][(g + hi * 8) * N2P + col] = sigmoidf_(acc2[n][g] + bias);
        }

        // ---- final logit: dot(h2[r,0:40], Wf) + bf, then mask ----
        if (lane < 16) {
            const int t = t0 + lane;
            if (t < T_) {
                float s = bfv;
                #pragma unroll
                for (int g2 = 0; g2 < H2_; ++g2)
                    s += h2_sh[w][lane * N2P + g2] * wf_sh[g2];
                logits_sh[t] = (mask[(size_t)b * T_ + t] == 0) ? CONST_MIN : s;
            }
        }
    }
    __syncthreads();

    // ---- block softmax over T=200 ----
    float lmax = -3.0e38f;
    for (int t = tid; t < T_; t += 128) lmax = fmaxf(lmax, logits_sh[t]);
    red_sh[tid] = lmax; __syncthreads();
    #pragma unroll
    for (int s = 64; s > 0; s >>= 1) {
        if (tid < s) red_sh[tid] = fmaxf(red_sh[tid], red_sh[tid + s]);
        __syncthreads();
    }
    const float mx = red_sh[0];
    __syncthreads();
    float lsum = 0.f;
    for (int t = tid; t < T_; t += 128) {
        const float e = __expf(logits_sh[t] - mx);
        logits_sh[t] = e;
        lsum += e;
    }
    red_sh[tid] = lsum; __syncthreads();
    #pragma unroll
    for (int s = 64; s > 0; s >>= 1) {
        if (tid < s) red_sh[tid] += red_sh[tid + s];
        __syncthreads();
    }
    const float inv = 1.f / red_sh[0];
    __syncthreads();

    // ---- weighted sum of v: two groups of 64 threads, 100 rows each ----
    {
        const int grp = tid >> 6, d = tid & 63;
        const float* vb = &v[((size_t)b * T_ + grp * 100) * D_ + d];
        float accv = 0.f;
        for (int t = 0; t < 100; ++t) {
            if ((t & 7) == 0) __builtin_prefetch(vb + (t + 16) * D_, 0, 1);
            accv += logits_sh[grp * 100 + t] * vb[(size_t)t * D_];
        }
        part_sh[grp][d] = accv;
    }
    __syncthreads();
    if (tid < D_)
        out[(size_t)b * D_ + tid] = (part_sh[0][tid] + part_sh[1][tid]) * inv;
}

// ---------------------------------------------------------------------------
extern "C" void kernel_launch(void* const* d_in, const int* in_sizes, int n_in,
                              void* d_out, int out_size, void* d_ws, size_t ws_size,
                              hipStream_t stream) {
    (void)in_sizes; (void)n_in; (void)out_size; (void)ws_size;
    const float* q    = (const float*)d_in[0];
    const float* k    = (const float*)d_in[1];
    const float* v    = (const float*)d_in[2];
    const int*   mask = (const int*)  d_in[3];
    const float* W1   = (const float*)d_in[4];
    const float* b1   = (const float*)d_in[5];
    const float* W2   = (const float*)d_in[6];
    const float* b2   = (const float*)d_in[7];
    const float* Wf   = (const float*)d_in[8];
    const float* bf   = (const float*)d_in[9];

    _Float16* w1m = (_Float16*)d_ws;            // 80*128 halves
    _Float16* w2h = w1m + H1_ * K1M;            // 48*96 halves
    float*    w1ac = (float*)(w2h + N2P * K2P); // 64*80 floats (offset is 4B-aligned)

    prep_kernel<<<1, 256, 0, stream>>>(W1, W2, w1m, w2h, w1ac);
    attn_mlp_kernel<<<B_, 128, 0, stream>>>(q, k, v, mask, b1, b2, Wf, bf,
                                            w1m, w2h, w1ac, (float*)d_out);
}
